// TrainableHedgehog_45165876085251
// MI455X (gfx1250) — compile-verified, matrix operand order checked by reading, not run on hardware
//
#include <hip/hip_runtime.h>
#include <hip/hip_bf16.h>
#include <math.h>

typedef __attribute__((ext_vector_type(16))) _Float16 v16h;
typedef __attribute__((ext_vector_type(8)))  float    v8f;

// Problem constants (from reference setup_inputs)
constexpr int B_  = 2;
constexpr int H_  = 16;
constexpr int L_  = 2048;
constexpr int D_  = 64;     // head dim
constexpr int F_  = 64;     // feature-map half width
constexpr int FF  = 2 * F_; // 128 features
constexpr int C_  = 64;     // chunk length
constexpr int NC  = L_ / C_;  // 32 chunks
constexpr int BH  = B_ * H_;  // 32

constexpr float EPS = 1e-12f;

// Workspace layout (bytes)
constexpr size_t PHI_BYTES = (size_t)BH * L_ * FF * sizeof(_Float16);        // 16 MB each
constexpr size_t SC_BYTES  = (size_t)BH * NC * FF * D_ * sizeof(float);      // 32 MB
constexpr size_t ZC_BYTES  = (size_t)BH * NC * FF * sizeof(float);           // 0.5 MB

// ---------------------------------------------------------------------------
// WMMA helpers (gfx1250, wave32).  A/B 16-bit fragment layout (ISA 7.12.2):
//   lane l (0..31): row/col = l%16 ; K elements = kBase + 8*(l/16) + {0..7}
//   packed in v16h elems 0..7, and +16 more in elems 8..15.
// Source rows must be contiguous f16 with stride `stride` halves.
// ---------------------------------------------------------------------------
__device__ __forceinline__ v16h load_frag(const _Float16* src, int rowBase,
                                          int kBase, int stride) {
  const int lane = threadIdx.x & 31;
  const int row  = rowBase + (lane & 15);
  const int kb   = kBase + ((lane >> 4) << 3);
  const _Float16* p = src + (size_t)row * stride + kb;
  v16h r;
#pragma unroll
  for (int i = 0; i < 8; ++i) r[i] = p[i];
#pragma unroll
  for (int i = 0; i < 8; ++i) r[8 + i] = p[16 + i];
  return r;
}

__device__ __forceinline__ v8f wmma32(v16h a, v16h b, v8f c) {
  return __builtin_amdgcn_wmma_f32_16x16x32_f16(false, a, false, b,
                                                (short)0, c, false, false);
}

// ---------------------------------------------------------------------------
// Pass A: feature maps (phi_q, phi_k -> f16) + per-chunk local KV state
//   Sc[f][d] = sum_t phiK[t][f] * v[t][d]   (f32, via WMMA)
//   zc[f]    = sum_t phiK[t][f]
// grid = BH*NC blocks, 256 threads (8 waves)
// ---------------------------------------------------------------------------
__global__ __launch_bounds__(256)
void hh_featmap_state(const float* __restrict__ q, const float* __restrict__ k,
                      const float* __restrict__ v, const float* __restrict__ Wq,
                      const float* __restrict__ Wk, const int* __restrict__ layer_idx,
                      _Float16* __restrict__ phiQ, _Float16* __restrict__ phiK,
                      float* __restrict__ Sc, float* __restrict__ zc) {
  const int blk = blockIdx.x;
  const int bh = blk / NC, c = blk % NC;
  const int h = bh % H_;
  const int layer = layer_idx[0];
  const int tid = threadIdx.x;
  const int lane = tid & 31, wave = tid >> 5;

  __shared__ float    W_lds[D_ * F_];   // 16 KB
  __shared__ float    x_lds[C_ * D_];   // 16 KB
  __shared__ float    o_lds[C_ * F_];   // 16 KB
  __shared__ _Float16 phiKT[FF * C_];   // 16 KB  [feature][time]
  __shared__ _Float16 vT[D_ * C_];      //  8 KB  [d][time]

  const size_t xbase = ((size_t)bh * L_ + (size_t)c * C_) * D_;

  for (int pass = 0; pass < 2; ++pass) {
    const float* W = (pass == 0 ? Wq : Wk) +
                     (((size_t)layer * H_ + h) * D_) * F_;
    const float* x = (pass == 0 ? q : k) + xbase;
    _Float16* phi = (pass == 0 ? phiQ : phiK) +
                    ((size_t)bh * L_ + (size_t)c * C_) * FF;

    for (int i = tid; i < D_ * F_; i += 256) W_lds[i] = W[i];
    for (int i = tid; i < C_ * D_; i += 256) x_lds[i] = x[i];
    __syncthreads();

    // o = x @ W  (64x64, f32 VALU — tiny)
    for (int i = tid; i < C_ * F_; i += 256) {
      const int r = i / F_, f = i % F_;
      float s = 0.f;
#pragma unroll 8
      for (int d = 0; d < D_; ++d) s += x_lds[r * D_ + d] * W_lds[d * F_ + f];
      o_lds[i] = s;
    }
    __syncthreads();

    // phi = [softmax(o), softmax(-o)] clipped at EPS, one row per thread
    if (tid < C_) {
      const int r = tid;
      float mx = -3.4e38f, mn = 3.4e38f;
      for (int f = 0; f < F_; ++f) {
        const float ov = o_lds[r * F_ + f];
        mx = fmaxf(mx, ov);
        mn = fminf(mn, ov);
      }
      float s1 = 0.f, s2 = 0.f;
      for (int f = 0; f < F_; ++f) {
        const float ov = o_lds[r * F_ + f];
        s1 += __expf(ov - mx);
        s2 += __expf(mn - ov);  // softmax(-o): max(-o) = -mn
      }
      const float i1 = 1.f / s1, i2 = 1.f / s2;
      for (int f = 0; f < F_; ++f) {
        const float ov = o_lds[r * F_ + f];
        const float p1 = fmaxf(__expf(ov - mx) * i1, EPS);
        const float p2 = fmaxf(__expf(mn - ov) * i2, EPS);
        phi[(size_t)r * FF + f]      = (_Float16)p1;
        phi[(size_t)r * FF + F_ + f] = (_Float16)p2;
        if (pass == 1) {
          phiKT[f * C_ + r]        = (_Float16)p1;
          phiKT[(F_ + f) * C_ + r] = (_Float16)p2;
        }
      }
    }
    __syncthreads();
  }

  // v chunk, transposed to [d][t] f16
  for (int i = tid; i < C_ * D_; i += 256) {
    const int t = i / D_, d = i % D_;
    vT[d * C_ + t] = (_Float16)v[xbase + i];
  }
  __syncthreads();

  // Sc = phiK^T @ v : [128 x 64] -> 8x4 = 32 tiles of 16x16, wave-uniform
  float* ScO = Sc + ((size_t)bh * NC + c) * FF * D_;
  for (int t = wave; t < 32; t += 8) {
    const int ti = t >> 2, tj = t & 3;
    v8f acc = {};
#pragma unroll
    for (int kb = 0; kb < C_; kb += 32) {
      v16h a = load_frag(phiKT, ti * 16, kb, C_);  // A[f][t]
      v16h b = load_frag(vT, tj * 16, kb, C_);     // B[t][d] = vT[d][t]
      acc = wmma32(a, b, acc);
    }
    const int n = lane & 15, mo = (lane >> 4) * 8;
#pragma unroll
    for (int r = 0; r < 8; ++r) {
      const int m = ti * 16 + r + mo;
      ScO[(size_t)m * D_ + tj * 16 + n] = acc[r];
    }
  }

  // zc[f] = column sums of phiK chunk
  if (tid < FF) {
    float s = 0.f;
    for (int t = 0; t < C_; ++t) s += (float)phiKT[tid * C_ + t];
    zc[((size_t)bh * NC + c) * FF + tid] = s;
  }
}

// ---------------------------------------------------------------------------
// Pass B: in-place EXCLUSIVE prefix over chunks of Sc and zc (per bh).
// grid = BH blocks, 256 threads
// ---------------------------------------------------------------------------
__global__ __launch_bounds__(256)
void hh_prefix(float* __restrict__ Sc, float* __restrict__ zc) {
  const int bh = blockIdx.x;
  const int tid = threadIdx.x;
  for (int e = tid; e < FF * D_; e += 256) {
    float run = 0.f;
    for (int c = 0; c < NC; ++c) {
      const size_t idx = ((size_t)bh * NC + c) * (FF * D_) + e;
      const float tmp = Sc[idx];
      Sc[idx] = run;
      run += tmp;
    }
  }
  if (tid < FF) {
    float run = 0.f;
    for (int c = 0; c < NC; ++c) {
      const size_t idx = ((size_t)bh * NC + c) * FF + tid;
      const float tmp = zc[idx];
      zc[idx] = run;
      run += tmp;
    }
  }
}

// ---------------------------------------------------------------------------
// Pass C: per chunk — masked intra scores (WMMA), row normalizer, and
//         out = (scores @ v + phiQ @ S_prefix) / den  (all WMMA, f32 acc)
// grid = BH*NC blocks, 256 threads (8 waves)
// ---------------------------------------------------------------------------
__global__ __launch_bounds__(256)
void hh_output(const float* __restrict__ v, const _Float16* __restrict__ phiQ,
               const _Float16* __restrict__ phiK, const float* __restrict__ Sc,
               const float* __restrict__ zc, float* __restrict__ out) {
  const int blk = blockIdx.x;
  const int bh = blk / NC, c = blk % NC;
  const int tid = threadIdx.x, lane = tid & 31, wave = tid >> 5;

  __shared__ _Float16 phiQ_l[C_ * FF];  // 16 KB
  __shared__ _Float16 phiK_l[C_ * FF];  // 16 KB
  __shared__ _Float16 vT[D_ * C_];      //  8 KB  [d][t]
  __shared__ _Float16 ST[D_ * FF];      // 16 KB  [d][f]  (state transposed)
  __shared__ _Float16 sc_l[C_ * C_];    //  8 KB  masked scores, row-major
  __shared__ float    invden[C_];
  __shared__ float    z_l[FF];

  const size_t phibase = ((size_t)bh * L_ + (size_t)c * C_) * FF;
  for (int i = tid; i < C_ * FF; i += 256) {
    phiQ_l[i] = phiQ[phibase + i];
    phiK_l[i] = phiK[phibase + i];
  }
  const size_t vbase = ((size_t)bh * L_ + (size_t)c * C_) * D_;
  for (int i = tid; i < C_ * D_; i += 256) {
    const int t = i / D_, d = i % D_;
    vT[d * C_ + t] = (_Float16)v[vbase + i];
  }
  const float* S = Sc + ((size_t)bh * NC + c) * FF * D_;
  for (int i = tid; i < FF * D_; i += 256) {
    const int f = i / D_, d = i % D_;
    ST[d * FF + f] = (_Float16)S[i];
  }
  if (tid < FF) z_l[tid] = zc[((size_t)bh * NC + c) * FF + tid];
  for (int i = tid; i < C_ * C_; i += 256) sc_l[i] = (_Float16)0.f;
  __syncthreads();

  // intra-chunk scores phiQ @ phiK^T, causal-masked, lower-triangle tiles only
  for (int t = wave; t < 16; t += 8) {
    const int ti = t >> 2, tj = t & 3;
    if (tj > ti) continue;  // wave-uniform
    v8f acc = {};
#pragma unroll
    for (int kb = 0; kb < FF; kb += 32) {
      v16h a = load_frag(phiQ_l, ti * 16, kb, FF);
      v16h b = load_frag(phiK_l, tj * 16, kb, FF);  // B[f][n] = phiK[n][f]
      acc = wmma32(a, b, acc);
    }
    const int n = lane & 15, mo = (lane >> 4) * 8;
#pragma unroll
    for (int r = 0; r < 8; ++r) {
      const int m = ti * 16 + r + mo;
      const int nn = tj * 16 + n;
      const float val = (m >= nn) ? acc[r] : 0.f;  // only bites on diag tiles
      sc_l[m * C_ + nn] = (_Float16)val;
    }
  }
  __syncthreads();

  // row normalizer: den = intra row-sum + phiQ . z_prefix + EPS
  if (tid < C_) {
    float s = 0.f;
    for (int n = 0; n < C_; ++n) s += (float)sc_l[tid * C_ + n];
    for (int f = 0; f < FF; ++f) s += (float)phiQ_l[tid * FF + f] * z_l[f];
    invden[tid] = 1.f / (s + EPS);
  }
  __syncthreads();

  // out = (scores @ v  +  phiQ @ S_prefix) * invden
  float* O = out + ((size_t)bh * L_ + (size_t)c * C_) * D_;
  for (int t = wave; t < 16; t += 8) {
    const int ti = t >> 2, tj = t & 3;
    v8f acc = {};
#pragma unroll
    for (int kb = 0; kb < C_; kb += 32) {        // intra: K over time
      v16h a = load_frag(sc_l, ti * 16, kb, C_);
      v16h b = load_frag(vT, tj * 16, kb, C_);
      acc = wmma32(a, b, acc);
    }
#pragma unroll
    for (int kb = 0; kb < FF; kb += 32) {        // inter: K over features
      v16h a = load_frag(phiQ_l, ti * 16, kb, FF);
      v16h b = load_frag(ST, tj * 16, kb, FF);
      acc = wmma32(a, b, acc);
    }
    const int n = lane & 15, mo = (lane >> 4) * 8;
#pragma unroll
    for (int r = 0; r < 8; ++r) {
      const int m = ti * 16 + r + mo;
      O[(size_t)m * D_ + tj * 16 + n] = acc[r] * invden[m];
    }
  }
}

// ---------------------------------------------------------------------------
extern "C" void kernel_launch(void* const* d_in, const int* in_sizes, int n_in,
                              void* d_out, int out_size, void* d_ws, size_t ws_size,
                              hipStream_t stream) {
  const float* q  = (const float*)d_in[0];
  const float* k  = (const float*)d_in[1];
  const float* v  = (const float*)d_in[2];
  const float* Wq = (const float*)d_in[3];
  const float* Wk = (const float*)d_in[4];
  const int* layer_idx = (const int*)d_in[5];

  char* ws = (char*)d_ws;
  _Float16* phiQ = (_Float16*)ws;
  _Float16* phiK = (_Float16*)(ws + PHI_BYTES);
  float* Sc = (float*)(ws + 2 * PHI_BYTES);
  float* zc = (float*)(ws + 2 * PHI_BYTES + SC_BYTES);

  float* out = (float*)d_out;

  hh_featmap_state<<<dim3(BH * NC), dim3(256), 0, stream>>>(
      q, k, v, Wq, Wk, layer_idx, phiQ, phiK, Sc, zc);
  hh_prefix<<<dim3(BH), dim3(256), 0, stream>>>(Sc, zc);
  hh_output<<<dim3(BH * NC), dim3(256), 0, stream>>>(v, phiQ, phiK, Sc, zc, out);
}
